// FusedLayerNormGRU_11854109737083
// MI455X (gfx1250) — matrix operand order repaired and checked by reading, not compiled, not run on hardware
//
#include <hip/hip_runtime.h>
#include <hip/hip_bf16.h>

typedef __bf16 bf16_t;
typedef __attribute__((ext_vector_type(16))) __bf16 v16bf;
typedef __attribute__((ext_vector_type(8)))  float  v8f;

struct U4 { unsigned int x, y, z, w; };      // POD 16-byte chunk
union FragU { U4 u[2]; v16bf v; };           // two b128 loads -> one WMMA operand

// ---------------------------------------------------------------------------
// f32 -> bf16 conversion, 4 elements per thread-step (all our sizes are /4)
// ---------------------------------------------------------------------------
struct B4 { bf16_t a, b, c, d; };            // 8-byte packed store

__global__ __launch_bounds__(256)
void cvt_f32_bf16(const float4* __restrict__ in, B4* __restrict__ out, int n4) {
  int i = blockIdx.x * blockDim.x + threadIdx.x;
  int stride = gridDim.x * blockDim.x;
  for (; i < n4; i += stride) {
    float4 f = in[i];
    B4 o;
    o.a = (bf16_t)f.x; o.b = (bf16_t)f.y; o.c = (bf16_t)f.z; o.d = (bf16_t)f.w;
    out[i] = o;
  }
}

// ---------------------------------------------------------------------------
// C[M,N] = A[M,K](bf16,row-major) * W[N,K](bf16,row-major)^T + bias[N]
// Templated wave tiling: wave computes TM x TN 16x16 tiles; WM x WN waves/WG.
// KSPLIT: grid.z K-partials written to C + z*cPartStride (bias only in z==0).
// A-fragment layout per ISA 7.12.2 (16-bit A 16x32):
//   lanes 0-15:  K = 0..7 (VGPR0-3), 16..23 (VGPR4-7)
//   lanes 16-31: K = 8..15, 24..31
// B-fragment (32x16): lane = column; same K order as A (lanes 0-15 K=0..7,16..23; lanes 16-31 K=8..15,24..31).
// ---------------------------------------------------------------------------
template <int TM, int TN, int WM, int WN, int KSPLIT>
__global__ __launch_bounds__(WM * WN * 32)
void wmma_gemm(const bf16_t* __restrict__ A, int lda,
               const bf16_t* __restrict__ W, int ldw,
               const float* __restrict__ bias,
               float* __restrict__ C, int ldc, int K, size_t cPartStride) {
  const int lane  = threadIdx.x & 31;
  const int wave  = threadIdx.x >> 5;
  const int waveM = wave / WN;
  const int waveN = wave % WN;
  const int half  = lane >> 4;
  const int l15   = lane & 15;

  const int m0 = blockIdx.y * (WM * TM * 16) + waveM * (TM * 16);
  const int n0 = blockIdx.x * (WN * TN * 16) + waveN * (TN * 16);

  const int ks = K / KSPLIT;
  const int k0 = (KSPLIT > 1) ? (int)blockIdx.z * ks : 0;
  float* Cz = C + (size_t)blockIdx.z * cPartStride;

  v8f acc[TM][TN] = {};

  const bf16_t* ap[TM];
  const bf16_t* bp[TN];
#pragma unroll
  for (int i = 0; i < TM; ++i)
    ap[i] = A + (size_t)(m0 + i * 16 + l15) * lda + k0 + half * 8;
#pragma unroll
  for (int j = 0; j < TN; ++j)
    bp[j] = W + (size_t)(n0 + j * 16 + l15) * ldw + k0 + half * 8;

  for (int kc = 0; kc < ks; kc += 32) {
    __builtin_prefetch(ap[0] + kc + 128, 0, 1);
    __builtin_prefetch(bp[0] + kc + 128, 0, 1);

    FragU a[TM], b[TN];
#pragma unroll
    for (int i = 0; i < TM; ++i) {
      a[i].u[0] = *(const U4*)(ap[i] + kc);
      a[i].u[1] = *(const U4*)(ap[i] + kc + 16);
    }
#pragma unroll
    for (int j = 0; j < TN; ++j) {
      b[j].u[0] = *(const U4*)(bp[j] + kc);
      b[j].u[1] = *(const U4*)(bp[j] + kc + 16);
    }
#pragma unroll
    for (int i = 0; i < TM; ++i)
#pragma unroll
      for (int j = 0; j < TN; ++j)
        acc[i][j] = __builtin_amdgcn_wmma_f32_16x16x32_bf16(
            false, a[i].v, false, b[j].v, (short)0, acc[i][j], false, false);
  }

  const bool addBias = (KSPLIT == 1) || (blockIdx.z == 0);
#pragma unroll
  for (int j = 0; j < TN; ++j) {
    const float bv = addBias ? bias[n0 + j * 16 + l15] : 0.0f;
#pragma unroll
    for (int i = 0; i < TM; ++i) {
      // C/D layout: VGPR q -> (M = q + 8*half, N = l15)
      float* c = Cz + (size_t)(m0 + i * 16 + half * 8) * ldc + n0 + j * 16 + l15;
#pragma unroll
      for (int q = 0; q < 8; ++q)
        c[(size_t)q * ldc] = acc[i][j][q] + bv;
    }
  }
}

// ---------------------------------------------------------------------------
// Per-step LayerNorm-GRU gate/update. One workgroup per batch row b.
// 256 threads = 8 wave32s; wave-level butterfly reductions + one LDS exchange.
// ---------------------------------------------------------------------------
__device__ __forceinline__ float wave_sum(float v) {
#pragma unroll
  for (int off = 16; off > 0; off >>= 1) v += __shfl_xor(v, off);
  return v;
}

__device__ __forceinline__
float2 ln_stats(float sum, float sq, float invH,
                float* s_sum, float* s_sq, int lane, int wid) {
  sum = wave_sum(sum);
  sq  = wave_sum(sq);
  __syncthreads();                        // protect s_* reuse across calls
  if (lane == 0) { s_sum[wid] = sum; s_sq[wid] = sq; }
  __syncthreads();
  float ts = 0.f, tq = 0.f;
#pragma unroll
  for (int w = 0; w < 8; ++w) { ts += s_sum[w]; tq += s_sq[w]; }
  float m   = ts * invH;
  float var = fmaxf(tq * invH - m * m, 0.0f);
  return make_float2(m, rsqrtf(var + 1e-5f));
}

__global__ __launch_bounds__(256)
void gru_gate_step(const float* __restrict__ xg_base,  // [B*T, 3H] (b_ih included)
                   const float* __restrict__ hg0,      // [B, 3H] K-partial 0 (b_hh incl.)
                   const float* __restrict__ hg1,      // [B, 3H] K-partial 1
                   float*       __restrict__ h,        // [B, H] state (in/out)
                   bf16_t*      __restrict__ h_bf,     // [B, H] bf16 mirror
                   const float* __restrict__ resid,    // [B, T, H]
                   float*       __restrict__ y,        // [B, T, H]
                   const float* __restrict__ lnw,      // [3, H]
                   const float* __restrict__ lnb,      // [3, H]
                   int t, int T, int H) {
  const int b    = blockIdx.x;
  const int tid  = threadIdx.x;
  const int lane = tid & 31;
  const int wid  = tid >> 5;
  const float invH = 1.0f / (float)H;

  const float* xg   = xg_base + ((size_t)b * T + t) * (size_t)(3 * H);
  const float* hga  = hg0 + (size_t)b * (3 * H);
  const float* hgb  = hg1 + (size_t)b * (3 * H);
  float*       hrow = h + (size_t)b * H;
  bf16_t*      hbr  = h_bf + (size_t)b * H;
  const float* rr   = resid + ((size_t)b * T + t) * H;
  float*       yr   = y + ((size_t)b * T + t) * H;

  __shared__ float s_sum[8];
  __shared__ float s_sq[8];

  float rg[4], zg[4], vb[4];

  // ---- r gate: LN(x_r + h_r) -> sigmoid -------------------------------
  float sum = 0.f, sq = 0.f;
  for (int j = tid, k = 0; j < H; j += 256, ++k) {
    float v = xg[j] + hga[j] + hgb[j];
    vb[k] = v; sum += v; sq += v * v;
  }
  float2 st = ln_stats(sum, sq, invH, s_sum, s_sq, lane, wid);
  for (int j = tid, k = 0; j < H; j += 256, ++k) {
    float v = (vb[k] - st.x) * st.y * lnw[j] + lnb[j];
    rg[k] = 1.0f / (1.0f + __expf(-v));
  }

  // ---- z gate: LN(x_z + h_z) -> sigmoid -------------------------------
  sum = 0.f; sq = 0.f;
  for (int j = tid, k = 0; j < H; j += 256, ++k) {
    float v = xg[H + j] + hga[H + j] + hgb[H + j];
    vb[k] = v; sum += v; sq += v * v;
  }
  st = ln_stats(sum, sq, invH, s_sum, s_sq, lane, wid);
  for (int j = tid, k = 0; j < H; j += 256, ++k) {
    float v = (vb[k] - st.x) * st.y * lnw[H + j] + lnb[H + j];
    zg[k] = 1.0f / (1.0f + __expf(-v));
  }

  // ---- n gate: LN(x_n + r*h_n) -> tanh; h_new = (1-z)*n + z*h ---------
  sum = 0.f; sq = 0.f;
  for (int j = tid, k = 0; j < H; j += 256, ++k) {
    float v = xg[2 * H + j] + rg[k] * (hga[2 * H + j] + hgb[2 * H + j]);
    vb[k] = v; sum += v; sq += v * v;
  }
  st = ln_stats(sum, sq, invH, s_sum, s_sq, lane, wid);
  for (int j = tid, k = 0; j < H; j += 256, ++k) {
    float n = tanhf((vb[k] - st.x) * st.y * lnw[2 * H + j] + lnb[2 * H + j]);
    float z = zg[k];
    float hn = (1.0f - z) * n + z * hrow[j];
    hrow[j] = hn;
    hbr[j]  = (bf16_t)hn;
    yr[j]   = hn + rr[j];
  }
}

// ---------------------------------------------------------------------------
extern "C" void kernel_launch(void* const* d_in, const int* in_sizes, int n_in,
                              void* d_out, int out_size, void* d_ws, size_t ws_size,
                              hipStream_t stream) {
  (void)in_sizes; (void)n_in; (void)out_size; (void)ws_size;
  const int B = 64, T = 512, D = 768, H = 768, L = 2;
  const int G = 3 * H;                      // 2304
  const size_t BT  = (size_t)B * T;         // 32768
  const size_t BTH = BT * H;

  const float* x    = (const float*)d_in[0];
  const float* h0   = (const float*)d_in[1];
  const float* w_ih = (const float*)d_in[2];
  const float* b_ih = (const float*)d_in[3];
  const float* w_hh = (const float*)d_in[4];
  const float* b_hh = (const float*)d_in[5];
  const float* ln_w = (const float*)d_in[6];
  const float* ln_b = (const float*)d_in[7];
  float* out = (float*)d_out;

  // workspace carve-up (256B aligned)
  char* p = (char*)d_ws;
  auto carve = [&](size_t bytes) {
    char* r = p; p += (bytes + 255) & ~(size_t)255; return r;
  };
  bf16_t* in_bf  = (bf16_t*)carve(BTH * sizeof(bf16_t));           // layer input, bf16
  bf16_t* wih_bf = (bf16_t*)carve((size_t)L * G * D * sizeof(bf16_t));
  bf16_t* whh_bf = (bf16_t*)carve((size_t)L * G * H * sizeof(bf16_t));
  float*  xgates = (float*)carve(BT * (size_t)G * sizeof(float));  // [B*T, 3H]
  float*  seq0   = (float*)carve(BTH * sizeof(float));             // layer-0 output
  float*  hg     = (float*)carve((size_t)2 * B * G * sizeof(float)); // 2 K-partials
  bf16_t* h_bf   = (bf16_t*)carve((size_t)B * H * sizeof(bf16_t)); // [B, H]

  float* h_state = out + BTH;   // [L, B, H] final hidden lives in d_out tail

  // h_state <- h0
  hipMemcpyAsync(h_state, h0, (size_t)L * B * H * sizeof(float),
                 hipMemcpyDeviceToDevice, stream);

  auto launch_cvt = [&](const float* src, bf16_t* dst, size_t n) {
    int n4 = (int)(n / 4);
    int blocks = (n4 + 256 * 4 - 1) / (256 * 4);
    if (blocks > 8192) blocks = 8192;
    if (blocks < 1) blocks = 1;
    cvt_f32_bf16<<<blocks, 256, 0, stream>>>((const float4*)src, (B4*)dst, n4);
  };

  // weights -> bf16 (once, both layers)
  launch_cvt(w_ih, wih_bf, (size_t)L * G * D);
  launch_cvt(w_hh, whh_bf, (size_t)L * G * H);

  for (int l = 0; l < L; ++l) {
    const float* layer_in = (l == 0) ? x : seq0;
    const float* residp   = (l == 0) ? x : seq0;
    float*       yout     = (l == 0) ? seq0 : out;

    // sequence input -> bf16, then big input-projection GEMM:
    // xgates[B*T, 3H] = layer_in @ w_ih[l]^T + b_ih[l]
    launch_cvt(layer_in, in_bf, BTH);
    {
      dim3 grid(G / 128, (unsigned)(BT / 64), 1);   // (18, 512)
      wmma_gemm<2, 2, 2, 4, 1><<<grid, 256, 0, stream>>>(
          in_bf, D, wih_bf + (size_t)l * G * D, D,
          b_ih + (size_t)l * G, xgates, G, D, 0);
    }

    // bf16 mirror of the hidden state for the recurrent GEMM
    launch_cvt(h_state + (size_t)l * B * H, h_bf, (size_t)B * H);

    dim3 sgrid(G / 64, 1, 2);                       // (36, 1, 2) split-K
    for (int t = 0; t < T; ++t) {
      // hg_z[B, 3H] = h @ w_hh[l][:, z*K/2:(z+1)*K/2]^T  (+ b_hh in z==0)
      wmma_gemm<1, 2, 4, 2, 2><<<sgrid, 256, 0, stream>>>(
          h_bf, H, whh_bf + (size_t)l * G * H, H,
          b_hh + (size_t)l * G, hg, G, H, (size_t)B * G);
      // gates, LayerNorms, state update, residual add
      gru_gate_step<<<B, 256, 0, stream>>>(
          xgates, hg, hg + (size_t)B * G,
          h_state + (size_t)l * B * H, h_bf,
          residp, yout,
          ln_w + (size_t)l * 3 * H, ln_b + (size_t)l * 3 * H,
          t, T, H);
    }
  }
}
